// PooledSelfAttention2d_84645215469688
// MI455X (gfx1250) — compile-verified, matrix operand order checked by reading, not run on hardware
//
#include <hip/hip_runtime.h>

// ---------------------------------------------------------------------------
// PooledSelfAttention2d for MI455X (gfx1250, wave32, WMMA).
// B=16, C=512, H=W=64, N=4096, M=1024.
// bf16 WMMA everywhere; flash-style fused softmax (beta never materialized);
// async global->LDS staging (ASYNCcnt) + TDM tensor_load_to_lds (TENSORcnt)
// for the V tile, both double-buffered.
// ---------------------------------------------------------------------------

typedef __attribute__((ext_vector_type(16))) __bf16 v16bf;
typedef __attribute__((ext_vector_type(8)))  __bf16 v8bf;
typedef __attribute__((ext_vector_type(4)))  __bf16 v4bf;
typedef __attribute__((ext_vector_type(2)))  __bf16 v2bf;
typedef __attribute__((ext_vector_type(8)))  float  v8f;
typedef __attribute__((ext_vector_type(4)))  int    v4i;
typedef __attribute__((ext_vector_type(4)))  unsigned int u32x4;
typedef __attribute__((ext_vector_type(8)))  int    i32x8;
typedef __attribute__((ext_vector_type(4)))  int    i32x4;

#if __has_builtin(__builtin_amdgcn_global_load_async_to_lds_b128) && \
    __has_builtin(__builtin_amdgcn_s_wait_asynccnt)
#define HAS_ASYNC 1
#else
#define HAS_ASYNC 0
#endif

#if __has_builtin(__builtin_amdgcn_tensor_load_to_lds) && \
    __has_builtin(__builtin_amdgcn_s_wait_tensorcnt)
#define HAS_TDM 1
#else
#define HAS_TDM 0
#endif

#if HAS_TDM
#define K_ASYNC_OUT 2     // only K copies ride ASYNCcnt (2 per wave per tile)
#else
#define K_ASYNC_OUT 10    // K (2) + V (8) copies per wave per tile
#endif

// Native bf16 converts (single v_cvt instructions on gfx1250; the compiler
// packs adjacent pairs into v_cvt_pk_bf16_f32).
__device__ __forceinline__ __bf16 f2bf(float f) { return (__bf16)f; }
__device__ __forceinline__ float  bf2f(__bf16 h) { return (float)h; }
__device__ __forceinline__ v2bf pk2bf(float a, float b) {
  v2bf r; r.x = (__bf16)a; r.y = (__bf16)b; return r;
}

// 16B/lane global->LDS copy: async (ASYNCcnt) when available, else direct.
__device__ __forceinline__ void cp16(const __bf16* g, __bf16* l) {
#if HAS_ASYNC
  __builtin_amdgcn_global_load_async_to_lds_b128(
      (__attribute__((address_space(1))) v4i*)(g),
      (__attribute__((address_space(3))) v4i*)(l), 0, 0);
#else
  *(v8bf*)l = *(const v8bf*)g;
#endif
}

#if HAS_TDM
__device__ __forceinline__ unsigned lds_off(void* p) {
  return (unsigned)(unsigned long long)(__attribute__((address_space(3))) char*)p;
}
// TDM descriptor for the V tile: 2D tensor gP[b] = 256 rows (c) x 1024 cols (m),
// tile = 64 (m, contiguous) x 256 (c), data_size 2B, LDS pad 4 DWORDs after
// every 32 DWORDs -> padded row stride of 72 halves (matches frag loader).
__device__ __forceinline__ void tdm_load_v(const __bf16* gtile, unsigned ldsOff) {
  unsigned long long ga = (unsigned long long)(size_t)gtile;
  u32x4 g0;
  g0[0] = 1u;                                            // count=1, no gather
  g0[1] = ldsOff;                                        // lds_addr (bytes)
  g0[2] = (unsigned)(ga & 0xFFFFFFFFu);                  // global_addr[31:0]
  g0[3] = (unsigned)((ga >> 32) & 0x01FFFFFFu)           // global_addr[56:32]
        | (2u << 30);                                    // type=2 ("image")
  i32x8 g1;
  g1[0] = (int)((1u << 16)      // data_size = 2 bytes
              | (1u << 20)      // pad_enable
              | (4u << 22)      // pad_interval: 32 DWORDs (one 128B tile row)
              | (3u << 25));    // pad_amount: 4 DWORDs (16B) -> 144B row pitch
  g1[1] = (int)(1024u << 16);   // tensor_dim0 = 1024 (bits 79:48, low half)
  g1[2] = (int)(256u << 16);    // tensor_dim0 hi=0 | tensor_dim1 = 256 (low)
  g1[3] = (int)(64u << 16);     // tensor_dim1 hi=0 | tile_dim0 = 64
  g1[4] = 256;                  // tile_dim1 = 256, tile_dim2 = 0
  g1[5] = 1024;                 // tensor_dim0_stride = 1024 (low 32)
  g1[6] = 0;                    // stride hi=0 | tensor_dim1_stride lo=0
  g1[7] = 0;
  i32x4 z4 = {0, 0, 0, 0};
  i32x8 z8 = {0, 0, 0, 0, 0, 0, 0, 0};
  // 6-arg form (clang-23 / therock-10.0 signature)
  __builtin_amdgcn_tensor_load_to_lds(g0, g1, z4, z4, z8, 0);
}
#endif

// Load a 16x32 bf16 A/B fragment from an LDS tile whose rows are `stride`
// halves apart (stride multiple of 8 -> 16B aligned chunks).
// CDNA5 ISA 7.12.2: lane l (idx=l&15 row/col, hi=l>>4):
//   halves 0..7  -> K = 8*hi .. 8*hi+7 ; halves 8..15 -> K = 16+8*hi ..
__device__ __forceinline__ v16bf frag_ab(const __bf16* tile, int row0, int stride) {
  int l   = (int)(threadIdx.x & 31u);
  int idx = l & 15;
  int hi  = l >> 4;
  const __bf16* p = tile + (size_t)(row0 + idx) * stride + 8 * hi;
  union { v16bf v; v8bf h[2]; } u;
  u.h[0] = *(const v8bf*)(p);
  u.h[1] = *(const v8bf*)(p + 16);
  return u.v;
}

#define WMMA_BF16(A, Bf, Cacc) \
  __builtin_amdgcn_wmma_f32_16x16x32_bf16(false, (A), false, (Bf), (short)0, (Cacc), false, false)

// ---------------------------------------------------------------------------
// Kernel 1: fused projection GEMM, software-pipelined.
//   Y(384x4096) = relu(Wcat(384x512) @ X(512x4096) + bcat)  per batch.
// ---------------------------------------------------------------------------
__global__ __launch_bounds__(256)
void proj_gemm_kernel(const float* __restrict__ x,
                      const float* __restrict__ Wt, const float* __restrict__ bt,
                      const float* __restrict__ Wp, const float* __restrict__ bp,
                      const float* __restrict__ Wg, const float* __restrict__ bg,
                      __bf16* __restrict__ thetaT,
                      __bf16* __restrict__ phiFull,
                      __bf16* __restrict__ gFull) {
  __shared__ __align__(16) __bf16 As[128 * 40];   // [m][k], k padded 32->40
  __shared__ __align__(16) __bf16 Bs[128 * 40];   // [n][k]
  const int b  = blockIdx.z;
  const int m0 = blockIdx.y * 128;
  const int n0 = blockIdx.x * 128;
  const int t  = threadIdx.x;
  const int wave = t >> 5;
  const int wy = wave >> 2;
  const int wx = wave & 3;
  const float* xb = x + (size_t)b * 512 * 4096;

  // K-invariant source pointers / LDS destinations (hoisted: no branching in loop)
  const float* aSrc[4]; __bf16* aDst[4];
  const float* bSrc[4]; int bRow[4], bCol[4];
#pragma unroll
  for (int i = 0; i < 4; ++i) {
    int cc = t + 256 * i;
    int r = cc >> 3, c4 = (cc & 7) * 4;
    int gm = m0 + r;
    const float* w = (gm < 64)  ? Wt + (size_t)gm * 512
                   : (gm < 128) ? Wp + (size_t)(gm - 64) * 512
                                : Wg + (size_t)(gm - 128) * 512;
    aSrc[i] = w + c4;
    aDst[i] = As + r * 40 + c4;
    int k = cc >> 5, n4 = (cc & 31) * 4;
    bSrc[i] = xb + (size_t)k * 4096 + n0 + n4;
    bRow[i] = n4; bCol[i] = k;
  }

  float4 aR[4], bR[4];
#pragma unroll
  for (int i = 0; i < 4; ++i) {
    aR[i] = *(const float4*)(aSrc[i]);
    bR[i] = *(const float4*)(bSrc[i]);
  }

  v8f acc[4][2] = {};

  for (int k0 = 0; k0 < 512; k0 += 32) {
#pragma unroll
    for (int i = 0; i < 4; ++i) {
      union { v4bf v; v2bf h[2]; } u;
      u.h[0] = pk2bf(aR[i].x, aR[i].y);
      u.h[1] = pk2bf(aR[i].z, aR[i].w);
      *(v4bf*)(aDst[i]) = u.v;
      Bs[(bRow[i] + 0) * 40 + bCol[i]] = f2bf(bR[i].x);
      Bs[(bRow[i] + 1) * 40 + bCol[i]] = f2bf(bR[i].y);
      Bs[(bRow[i] + 2) * 40 + bCol[i]] = f2bf(bR[i].z);
      Bs[(bRow[i] + 3) * 40 + bCol[i]] = f2bf(bR[i].w);
    }
    __syncthreads();
    if (k0 + 32 < 512) {                 // prefetch next K-slab during WMMAs
#pragma unroll
      for (int i = 0; i < 4; ++i) {
        aR[i] = *(const float4*)(aSrc[i] + k0 + 32);
        bR[i] = *(const float4*)(bSrc[i] + (size_t)(k0 + 32) * 4096);
      }
    }
#pragma unroll
    for (int mt = 0; mt < 4; ++mt) {
      v16bf a = frag_ab(As, wy * 64 + mt * 16, 40);
#pragma unroll
      for (int nt = 0; nt < 2; ++nt) {
        v16bf bf = frag_ab(Bs, wx * 32 + nt * 16, 40);
        acc[mt][nt] = WMMA_BF16(a, bf, acc[mt][nt]);
      }
    }
    __syncthreads();
  }

  // Epilogue: 16-row tile lies entirely in one output region -> uniform branch.
  const int lane = t & 31, col = lane & 15, hi = lane >> 4;
#pragma unroll
  for (int mt = 0; mt < 4; ++mt) {
    const int gmBase = m0 + wy * 64 + mt * 16;
    if (gmBase < 64) {
#pragma unroll
      for (int nt = 0; nt < 2; ++nt)
#pragma unroll
        for (int r = 0; r < 8; ++r) {
          int gm = gmBase + r + 8 * hi;
          int gn = n0 + wx * 32 + nt * 16 + col;
          float v = fmaxf(acc[mt][nt][r] + bt[gm], 0.f);
          thetaT[((size_t)b * 4096 + gn) * 64 + gm] = f2bf(v);
        }
    } else if (gmBase < 128) {
#pragma unroll
      for (int nt = 0; nt < 2; ++nt)
#pragma unroll
        for (int r = 0; r < 8; ++r) {
          int gm = gmBase + r + 8 * hi;
          int gn = n0 + wx * 32 + nt * 16 + col;
          float v = fmaxf(acc[mt][nt][r] + bp[gm - 64], 0.f);
          phiFull[((size_t)b * 64 + (gm - 64)) * 4096 + gn] = f2bf(v);
        }
    } else {
#pragma unroll
      for (int nt = 0; nt < 2; ++nt)
#pragma unroll
        for (int r = 0; r < 8; ++r) {
          int gm = gmBase + r + 8 * hi;
          int gn = n0 + wx * 32 + nt * 16 + col;
          float v = fmaxf(acc[mt][nt][r] + bg[gm - 128], 0.f);
          gFull[((size_t)b * 256 + (gm - 128)) * 4096 + gn] = f2bf(v);
        }
    }
  }
}

// ---------------------------------------------------------------------------
// Kernel 2: 2x2 maxpool.  phiT[b][m][c] (m-major) and gP[b][c][m] (c-major).
// ---------------------------------------------------------------------------
__global__ __launch_bounds__(256)
void pool_kernel(const __bf16* __restrict__ phiFull, const __bf16* __restrict__ gFull,
                 __bf16* __restrict__ phiT, __bf16* __restrict__ gP) {
  const size_t PHI_N = (size_t)16 * 1024 * 64;
  const size_t G_N   = (size_t)16 * 256 * 1024;
  size_t idx = (size_t)blockIdx.x * 256 + threadIdx.x;
  if (idx < PHI_N) {
    int c = (int)(idx & 63);
    size_t tmp = idx >> 6;
    int m = (int)(tmp & 1023);
    int b = (int)(tmp >> 10);
    int n = (m >> 5) * 128 + (m & 31) * 2;
    const __bf16* s = phiFull + ((size_t)b * 64 + c) * 4096 + n;
    float v = fmaxf(fmaxf(bf2f(s[0]), bf2f(s[1])), fmaxf(bf2f(s[64]), bf2f(s[65])));
    phiT[idx] = f2bf(v);
  } else if (idx < PHI_N + G_N) {
    size_t j = idx - PHI_N;
    int m = (int)(j & 1023);
    size_t tmp = j >> 10;
    int c = (int)(tmp & 255);
    int b = (int)(tmp >> 8);
    int n = (m >> 5) * 128 + (m & 31) * 2;
    const __bf16* s = gFull + ((size_t)b * 256 + c) * 4096 + n;
    float v = fmaxf(fmaxf(bf2f(s[0]), bf2f(s[1])), fmaxf(bf2f(s[64]), bf2f(s[65])));
    gP[j] = f2bf(v);
  }
}

// ---------------------------------------------------------------------------
// Kernel 3: flash attention; K via async copies, V via TDM, double-buffered.
// Per workgroup: 128 queries of one batch, 16 KV tiles of 64.
// ---------------------------------------------------------------------------
__global__ __launch_bounds__(256)
void attn_kernel(const __bf16* __restrict__ thetaT, const __bf16* __restrict__ phiT,
                 const __bf16* __restrict__ gP, __bf16* __restrict__ OT) {
  __shared__ __align__(16) __bf16 Qs[128 * 72];
  __shared__ __align__(16) __bf16 Ks[2][64 * 72];
  __shared__ __align__(16) __bf16 Vs[2][256 * 72];
  __shared__ __align__(16) float  Sb[128 * 68];
  __shared__ __align__(16) __bf16 Pb[128 * 72];
  __shared__ float mrow[128], lrow[128], arow[128];

  const int bb = blockIdx.y;
  const int q0 = blockIdx.x * 128;
  const int t  = threadIdx.x;
  const int wave = t >> 5;
  const int wq = wave >> 2;
  const int wc = wave & 3;
  const int lane = t & 31, col = lane & 15, hi = lane >> 4;

  // stage Q (async): 4 chunks/thread
#pragma unroll
  for (int i = 0; i < 4; ++i) {
    int cc = t + 256 * i;
    int r  = cc >> 3;
    int c8 = (cc & 7) * 8;
    cp16(thetaT + ((size_t)bb * 4096 + q0 + r) * 64 + c8, Qs + r * 72 + c8);
  }
  if (t < 128) { mrow[t] = -1e30f; lrow[t] = 0.f; }

  auto issueKV = [&](int it, int buf) {
    const int m0 = it * 64;
#pragma unroll
    for (int i = 0; i < 2; ++i) {           // K tile: 2 async copies per wave
      int cc = t + 256 * i;
      int r  = cc >> 3;
      int c8 = (cc & 7) * 8;
      cp16(phiT + ((size_t)bb * 1024 + m0 + r) * 64 + c8, &Ks[buf][r * 72 + c8]);
    }
#if HAS_TDM
    if (wave == 0)                          // V tile: one TDM DMA per tile
      tdm_load_v(gP + (size_t)bb * 256 * 1024 + m0, lds_off(&Vs[buf][0]));
#else
#pragma unroll
    for (int i = 0; i < 8; ++i) {
      int cc = t + 256 * i;
      int c  = cc >> 3;
      int j8 = (cc & 7) * 8;
      cp16(gP + ((size_t)bb * 256 + c) * 1024 + m0 + j8, &Vs[buf][c * 72 + j8]);
    }
#endif
  };

  v8f acc[4][4] = {};
  issueKV(0, 0);
  int cur = 0;

  for (int it = 0; it < 16; ++it) {
    if (it + 1 < 16) {
      issueKV(it + 1, cur ^ 1);    // stream next tile while computing this one
#if HAS_ASYNC
      __builtin_amdgcn_s_wait_asynccnt(K_ASYNC_OUT);
#endif
#if HAS_TDM
      if (wave == 0) __builtin_amdgcn_s_wait_tensorcnt(1);
#endif
    } else {
#if HAS_ASYNC
      __builtin_amdgcn_s_wait_asynccnt(0);
#endif
#if HAS_TDM
      if (wave == 0) __builtin_amdgcn_s_wait_tensorcnt(0);
#endif
    }
    __syncthreads();
    const __bf16* K = Ks[cur];
    const __bf16* V = Vs[cur];

    // S = Q @ K^T : rows wq*64+mt*16, cols wc*16, K-dim 64 channels
#pragma unroll
    for (int mt = 0; mt < 4; ++mt) {
      v8f s = {};
#pragma unroll
      for (int ks = 0; ks < 64; ks += 32) {
        v16bf a  = frag_ab(Qs + ks, wq * 64 + mt * 16, 72);
        v16bf bf = frag_ab(K + ks, wc * 16, 72);
        s = WMMA_BF16(a, bf, s);
      }
#pragma unroll
      for (int r = 0; r < 8; ++r)
        Sb[(wq * 64 + mt * 16 + r + 8 * hi) * 68 + wc * 16 + col] = s[r];
    }
    __syncthreads();

    // online softmax: one thread per query row
    if (t < 128) {
      const float* srow = Sb + t * 68;
      float tm = -1e30f;
#pragma unroll
      for (int j = 0; j < 16; ++j) {
        float4 v = *(const float4*)(srow + j * 4);
        tm = fmaxf(tm, fmaxf(fmaxf(v.x, v.y), fmaxf(v.z, v.w)));
      }
      float mold = mrow[t];
      float mnew = fmaxf(mold, tm);
      float alpha = __expf(mold - mnew);
      float sum = 0.f;
      __bf16* prow = Pb + t * 72;
#pragma unroll
      for (int j = 0; j < 64; j += 2) {
        float p0 = __expf(srow[j] - mnew);
        float p1 = __expf(srow[j + 1] - mnew);
        sum += p0 + p1;
        *(v2bf*)(prow + j) = pk2bf(p0, p1);
      }
      lrow[t] = lrow[t] * alpha + sum;
      mrow[t] = mnew;
      arow[t] = alpha;
    }
    __syncthreads();

    // rescale accumulators, then O += P @ V
#pragma unroll
    for (int mt = 0; mt < 4; ++mt) {
#pragma unroll
      for (int r = 0; r < 8; ++r) {
        float al = arow[wq * 64 + mt * 16 + r + 8 * hi];
#pragma unroll
        for (int ct = 0; ct < 4; ++ct) acc[mt][ct][r] *= al;
      }
    }
#pragma unroll
    for (int mt = 0; mt < 4; ++mt) {
#pragma unroll
      for (int ks = 0; ks < 64; ks += 32) {
        v16bf a = frag_ab(Pb + ks, wq * 64 + mt * 16, 72);
#pragma unroll
        for (int ct = 0; ct < 4; ++ct) {
          v16bf bf = frag_ab(V + ks, wc * 64 + ct * 16, 72);
          acc[mt][ct] = WMMA_BF16(a, bf, acc[mt][ct]);
        }
      }
    }
    __syncthreads();   // protects Vs[cur] before it is refilled at it+2
    cur ^= 1;
  }

  // epilogue: divide by row sums, store OT[b][n][c]
#pragma unroll
  for (int mt = 0; mt < 4; ++mt) {
#pragma unroll
    for (int r = 0; r < 8; ++r) {
      int qloc = wq * 64 + mt * 16 + r + 8 * hi;
      float inv = 1.f / lrow[qloc];
      size_t base = ((size_t)bb * 4096 + q0 + qloc) * 256;
#pragma unroll
      for (int ct = 0; ct < 4; ++ct)
        OT[base + wc * 64 + ct * 16 + col] = f2bf(acc[mt][ct][r] * inv);
    }
  }
}

// ---------------------------------------------------------------------------
// Kernel 4: out = gamma * relu(W_o(512x256) @ O(256x4096) + b_o) + x,
// software-pipelined.
// ---------------------------------------------------------------------------
__global__ __launch_bounds__(256)
void out_gemm_kernel(const __bf16* __restrict__ OT,
                     const float* __restrict__ Wo, const float* __restrict__ bo,
                     const float* __restrict__ gamma, const float* __restrict__ x,
                     float* __restrict__ out) {
  __shared__ __align__(16) __bf16 As[128 * 40];
  __shared__ __align__(16) __bf16 Bs[128 * 40];
  const int b  = blockIdx.z;
  const int m0 = blockIdx.y * 128;
  const int n0 = blockIdx.x * 128;
  const int t  = threadIdx.x;
  const int wave = t >> 5;
  const int wy = wave >> 2;
  const int wx = wave & 3;

  const float* aSrc[4]; __bf16* aDst[4];
  const __bf16* bSrc[2]; __bf16* bDst[2];
#pragma unroll
  for (int i = 0; i < 4; ++i) {
    int cc = t + 256 * i;
    int r = cc >> 3, c4 = (cc & 7) * 4;
    aSrc[i] = Wo + (size_t)(m0 + r) * 256 + c4;
    aDst[i] = As + r * 40 + c4;
  }
#pragma unroll
  for (int i = 0; i < 2; ++i) {
    int cc = t + 256 * i;
    int r = cc >> 2, c8 = (cc & 3) * 8;
    bSrc[i] = OT + ((size_t)b * 4096 + n0 + r) * 256 + c8;
    bDst[i] = Bs + r * 40 + c8;
  }

  float4 aR[4]; v8bf bR[2];
#pragma unroll
  for (int i = 0; i < 4; ++i) aR[i] = *(const float4*)(aSrc[i]);
#pragma unroll
  for (int i = 0; i < 2; ++i) bR[i] = *(const v8bf*)(bSrc[i]);

  v8f acc[4][2] = {};

  for (int k0 = 0; k0 < 256; k0 += 32) {
#pragma unroll
    for (int i = 0; i < 4; ++i) {
      union { v4bf v; v2bf h[2]; } u;
      u.h[0] = pk2bf(aR[i].x, aR[i].y);
      u.h[1] = pk2bf(aR[i].z, aR[i].w);
      *(v4bf*)(aDst[i]) = u.v;
    }
#pragma unroll
    for (int i = 0; i < 2; ++i) *(v8bf*)(bDst[i]) = bR[i];
    __syncthreads();
    if (k0 + 32 < 256) {
#pragma unroll
      for (int i = 0; i < 4; ++i) aR[i] = *(const float4*)(aSrc[i] + k0 + 32);
#pragma unroll
      for (int i = 0; i < 2; ++i) bR[i] = *(const v8bf*)(bSrc[i] + k0 + 32);
    }
#pragma unroll
    for (int mt = 0; mt < 4; ++mt) {
      v16bf a = frag_ab(As, wy * 64 + mt * 16, 40);
#pragma unroll
      for (int nt = 0; nt < 2; ++nt) {
        v16bf bf = frag_ab(Bs, wx * 32 + nt * 16, 40);
        acc[mt][nt] = WMMA_BF16(a, bf, acc[mt][nt]);
      }
    }
    __syncthreads();
  }

  const float gmma = gamma[0];
  const int lane = t & 31, col = lane & 15, hi = lane >> 4;
#pragma unroll
  for (int mt = 0; mt < 4; ++mt) {
#pragma unroll
    for (int nt = 0; nt < 2; ++nt) {
#pragma unroll
      for (int r = 0; r < 8; ++r) {
        int gm = m0 + wy * 64 + mt * 16 + r + 8 * hi;
        int gn = n0 + wx * 32 + nt * 16 + col;
        float v = fmaxf(acc[mt][nt][r] + bo[gm], 0.f);
        size_t off = ((size_t)b * 512 + gm) * 4096 + gn;
        out[off] = gmma * v + x[off];
      }
    }
  }
}

// ---------------------------------------------------------------------------
// Launcher
// ---------------------------------------------------------------------------
extern "C" void kernel_launch(void* const* d_in, const int* in_sizes, int n_in,
                              void* d_out, int out_size, void* d_ws, size_t ws_size,
                              hipStream_t stream) {
  const float* x  = (const float*)d_in[0];
  const float* Wt = (const float*)d_in[1];
  const float* bt = (const float*)d_in[2];
  const float* Wp = (const float*)d_in[3];
  const float* bp = (const float*)d_in[4];
  const float* Wg = (const float*)d_in[5];
  const float* bg = (const float*)d_in[6];
  const float* Wo = (const float*)d_in[7];
  const float* bo = (const float*)d_in[8];
  const float* gm = (const float*)d_in[9];
  float* out = (float*)d_out;

  char* ws = (char*)d_ws;
  __bf16* thetaT  = (__bf16*)(ws + 0);           //  8 MB: 16*4096*64
  __bf16* phiFull = (__bf16*)(ws + 8388608);     //  8 MB: 16*64*4096
  __bf16* gFull   = (__bf16*)(ws + 16777216);    // 32 MB: 16*256*4096
  __bf16* phiT    = (__bf16*)(ws + 50331648);    //  2 MB: 16*1024*64
  __bf16* gP      = (__bf16*)(ws + 52428800);    //  8 MB: 16*256*1024
  __bf16* OT      = (__bf16*)(ws + 60817408);    // 32 MB: 16*4096*256

  proj_gemm_kernel<<<dim3(32, 3, 16), 256, 0, stream>>>(
      x, Wt, bt, Wp, bp, Wg, bg, thetaT, phiFull, gFull);

  const size_t pool_total = (size_t)16 * 1024 * 64 + (size_t)16 * 256 * 1024;
  pool_kernel<<<dim3((unsigned)((pool_total + 255) / 256), 1, 1), 256, 0, stream>>>(
      phiFull, gFull, phiT, gP);

  attn_kernel<<<dim3(32, 16, 1), 256, 0, stream>>>(thetaT, phiT, gP, OT);

  out_gemm_kernel<<<dim3(32, 4, 16), 256, 0, stream>>>(OT, Wo, bo, gm, x, out);
}
